// Codebook_25778393710732
// MI455X (gfx1250) — matrix operand order repaired and checked by reading, not compiled, not run on hardware
//
#include <hip/hip_runtime.h>
#include <hip/hip_bf16.h>

// ---------------------------------------------------------------------------
// VQ-VAE codebook quantization for MI455X (gfx1250, wave32, WMMA).
//   z:         [16,256,64,64] fp32
//   embedding: [1024,256]     fp32
// Outputs (concatenated in d_out as float): z_q (16.7M), idx (65536), loss (1)
// ---------------------------------------------------------------------------

typedef __attribute__((ext_vector_type(16))) __bf16 bf16x16;
typedef __attribute__((ext_vector_type(8)))  __bf16 bf16x8;
typedef __attribute__((ext_vector_type(8)))  float  v8f;

#define B_DIM 16
#define C_DIM 256
#define H_DIM 64
#define W_DIM 64
#define NROWS 65536          // B*H*W
#define KCODES 1024
#define CHUNK 32             // codes per LDS buffer (double-buffered)
#define WPITCH 264           // padded ushorts per LDS code row (8*33 -> 16B aligned, bank-spread)
#define ZQ_ELEMS 16777216

__device__ __forceinline__ unsigned short f32_to_bf16(float f) {
    unsigned int u = __float_as_uint(f);
    unsigned int r = u + 0x7FFFu + ((u >> 16) & 1u);   // round-to-nearest-even
    return (unsigned short)(r >> 16);
}

__device__ __forceinline__ bf16x16 cat16(bf16x8 a, bf16x8 b) {
    return __builtin_shufflevector(a, b, 0,1,2,3,4,5,6,7,8,9,10,11,12,13,14,15);
}

// ---------------------------------------------------------------------------
__global__ void init_acc(float* acc) { *acc = 0.0f; }

// ---------------------------------------------------------------------------
// Normalize codebook rows -> bf16, plus 0.5*||wn||^2 per code.
__global__ __launch_bounds__(256) void prep_w(const float* __restrict__ emb,
                                              unsigned short* __restrict__ wn,
                                              float* __restrict__ half_wsq) {
    __shared__ float red[256];
    const int k = blockIdx.x;
    const int t = threadIdx.x;
    const float x = emb[k * C_DIM + t];
    red[t] = x * x;
    __syncthreads();
    for (int s = 128; s > 0; s >>= 1) {
        if (t < s) red[t] += red[t + s];
        __syncthreads();
    }
    const float sum = red[0];
    const float iv = 1.0f / fmaxf(sqrtf(sum), 1e-12f);
    wn[k * C_DIM + t] = f32_to_bf16(x * iv);
    if (t == 0) half_wsq[k] = 0.5f * sum * iv * iv;
}

// ---------------------------------------------------------------------------
// W-axis L2-normalize z, transpose [B,C,H,W] -> zf[N,C] bf16 row-major.
__global__ __launch_bounds__(256) void prep_z(const float* __restrict__ z,
                                              unsigned short* __restrict__ zf,
                                              float* __restrict__ inv_norm) {
    __shared__ float tile[64][65];
    __shared__ float inv[64];
    const int blk  = blockIdx.x;           // 0..4095
    const int cblk = blk & 3;
    const int h    = (blk >> 2) & 63;
    const int b    = blk >> 8;
    const int t    = threadIdx.x;
    const int col  = t & 63;
    const int r4   = t >> 6;
    const int cbase = cblk * 64;

#pragma unroll
    for (int i = 0; i < 16; ++i) {
        const int r = r4 + 4 * i;          // local channel
        tile[r][col] = z[(((b * C_DIM + cbase + r) * H_DIM + h) * W_DIM) + col];
    }
    __syncthreads();
    if (t < 64) {
        float s = 0.0f;
        for (int w = 0; w < 64; ++w) { const float x = tile[t][w]; s += x * x; }
        const float iv = 1.0f / fmaxf(sqrtf(s), 1e-12f);
        inv[t] = iv;
        inv_norm[(b * C_DIM + cbase + t) * H_DIM + h] = iv;
    }
    __syncthreads();
#pragma unroll
    for (int i = 0; i < 16; ++i) {
        const int w = r4 + 4 * i;
        const int n = (b * H_DIM + h) * W_DIM + w;
        zf[n * C_DIM + cbase + col] = f32_to_bf16(tile[col][w] * inv[col]);   // col = local channel
    }
}

// ---------------------------------------------------------------------------
// Fused bf16 WMMA GEMM (zf @ wn^T) + argmax over 1024 codes.
// Block = 128 threads (4 waves). Each wave owns one 16-row tile of zf.
// Codebook streamed via double-buffered GLOBAL_LOAD_ASYNC_TO_LDS_B128,
// overlapping global fetch of chunk i+1 with WMMA compute of chunk i.
__global__ __launch_bounds__(128) void gemm_argmax(const unsigned short* __restrict__ zf,
                                                   const unsigned short* __restrict__ wn,
                                                   const float* __restrict__ half_wsq,
                                                   int* __restrict__ ws_idx,
                                                   float* __restrict__ out_idx) {
    __shared__ unsigned short lds_w[2][CHUNK * WPITCH];   // 2 x 16896 B
    __shared__ float red_val[4][256];
    __shared__ int   red_idx[4][256];

    const int t    = threadIdx.x;
    const int lane = t & 31;
    const int wave = t >> 5;
    const int hi   = lane >> 4;                  // half-wave select
    const int m0   = blockIdx.x * 64 + wave * 16;
    const int row  = m0 + (lane & 15);

    // --- Load A tile (16 rows x 256 c) per ISA 16-bit A layout:
    //     lane<16: elems 0..7 = K 0..7, 8..15 = K 16..23 (of each 32-chunk); lane>=16 shifted by 8.
    bf16x16 a[8];
    const unsigned short* arow = zf + (size_t)row * C_DIM;
#pragma unroll
    for (int s = 0; s < 8; ++s) {
        const bf16x8 lo = *(const bf16x8*)(arow + 32 * s + hi * 8);
        const bf16x8 h8 = *(const bf16x8*)(arow + 32 * s + 16 + hi * 8);
        a[s] = cat16(lo, h8);
    }

    float best[8];
    int   bidx[8];
#pragma unroll
    for (int r = 0; r < 8; ++r) { best[r] = -3.4e38f; bidx[r] = 0; }

    // Async stage of one 32-code chunk: 32x256 bf16 = 1024 b128 transfers,
    // 8 per thread, written straight into (padded) LDS, tracked by ASYNCcnt.
    auto issue_chunk = [&](int cb, int buf) {
        const unsigned lds_base = (unsigned)(size_t)(&lds_w[buf][0]);
#pragma unroll
        for (int i = 0; i < 8; ++i) {
            const int v    = t + 128 * i;        // vec8 index 0..1023
            const int code = v >> 5;             // 32 vec8 per code row
            const int c8   = (v & 31) * 8;
            const unsigned long long g =
                (unsigned long long)(size_t)(wn + (size_t)(cb + code) * C_DIM + c8);
            const unsigned loff = lds_base + (unsigned)(code * WPITCH + c8) * 2u;
            asm volatile("global_load_async_to_lds_b128 %0, %1, off"
                         :: "v"(loff), "v"(g) : "memory");
        }
    };

    issue_chunk(0, 0);
    int buf = 0;
    for (int cb = 0; cb < KCODES; cb += CHUNK) {
        asm volatile("s_wait_asynccnt 0" ::: "memory");   // my chunk-cb loads landed
        __syncthreads();                                  // everyone's landed; prev reads done
        if (cb + CHUNK < KCODES) issue_chunk(cb + CHUNK, buf ^ 1);

#pragma unroll
        for (int j = 0; j < 2; ++j) {            // 2 code-tiles of 16 per chunk
            const int codeN = cb + 16 * j + (lane & 15);
            const unsigned short* bcol = &lds_w[buf][(16 * j + (lane & 15)) * WPITCH + hi * 8];
            // Preload all 8 B fragments so ds_loads issue back-to-back ahead of the WMMA chain.
            bf16x16 bm[8];
#pragma unroll
            for (int s = 0; s < 8; ++s) {
                const bf16x8 blo = *(const bf16x8*)(bcol + 32 * s);
                const bf16x8 bhi = *(const bf16x8*)(bcol + 32 * s + 16);
                bm[s] = cat16(blo, bhi);
            }
            v8f acc = {0.f, 0.f, 0.f, 0.f, 0.f, 0.f, 0.f, 0.f};
#pragma unroll
            for (int s = 0; s < 8; ++s) {
                acc = __builtin_amdgcn_wmma_f32_16x16x32_bf16(
                          false, a[s], false, bm[s], (short)0, acc, false, false);
            }
            const float w = half_wsq[codeN];
#pragma unroll
            for (int r = 0; r < 8; ++r) {
                const float sc = acc[r] - w;     // argmax(sc) == argmin(distance)
                if (sc > best[r]) { best[r] = sc; bidx[r] = codeN; }   // strict > keeps first
            }
        }
        buf ^= 1;
    }

    // Cross-lane argmax reduce: (lane,r) owns M = r + 8*hi, slot = lane&15.
    __syncthreads();
#pragma unroll
    for (int r = 0; r < 8; ++r) {
        const int M = r + 8 * hi;
        red_val[wave][M * 16 + (lane & 15)] = best[r];
        red_idx[wave][M * 16 + (lane & 15)] = bidx[r];
    }
    __syncthreads();
    if (lane < 16) {
        float bv = red_val[wave][lane * 16];
        int   bi = red_idx[wave][lane * 16];
        for (int s = 1; s < 16; ++s) {
            const float v = red_val[wave][lane * 16 + s];
            const int  ix = red_idx[wave][lane * 16 + s];
            if (v > bv || (v == bv && ix < bi)) { bv = v; bi = ix; }   // tie -> smaller index
        }
        const int n = m0 + lane;
        ws_idx[n]  = bi;
        out_idx[n] = (float)bi;
    }
}

// ---------------------------------------------------------------------------
// Gather embedding[idx] -> z_q in [B,C,H,W]; accumulate 1.25*mean((z_q-zp)^2).
__global__ __launch_bounds__(256) void epilogue(const float* __restrict__ z,
                                                const float* __restrict__ emb,
                                                const float* __restrict__ inv_norm,
                                                const int* __restrict__ ws_idx,
                                                float* __restrict__ out_zq,
                                                float* __restrict__ loss_acc) {
    __shared__ float tile[64][65];
    __shared__ int   lidx[64];
    __shared__ float red[256];
    const int blk  = blockIdx.x;
    const int cblk = blk & 3;
    const int h    = (blk >> 2) & 63;
    const int b    = blk >> 8;
    const int t    = threadIdx.x;
    const int col  = t & 63;
    const int r4   = t >> 6;
    const int cbase = cblk * 64;
    const int nbase = (b * H_DIM + h) * W_DIM;

    if (t < 64) lidx[t] = ws_idx[nbase + t];
#pragma unroll
    for (int i = 0; i < 16; ++i) {
        const int r = r4 + 4 * i;
        const float iv = inv_norm[(b * C_DIM + cbase + r) * H_DIM + h];
        tile[r][col] = z[(((b * C_DIM + cbase + r) * H_DIM + h) * W_DIM) + col] * iv;  // zp
    }
    __syncthreads();

    float acc = 0.0f;
#pragma unroll
    for (int i = 0; i < 16; ++i) {
        const int w = r4 + 4 * i;                              // col = local channel
        const float e  = emb[(size_t)lidx[w] * C_DIM + cbase + col];  // coalesced over c
        const float zp = tile[col][w];
        const float d  = e - zp;
        acc += d * d;
        tile[col][w] = e;                                      // stash z_q (slot touched once)
    }
    __syncthreads();
#pragma unroll
    for (int i = 0; i < 16; ++i) {
        const int r = r4 + 4 * i;                              // local channel; col = w
        out_zq[(((b * C_DIM + cbase + r) * H_DIM + h) * W_DIM) + col] = tile[r][col];
    }

    red[t] = acc;
    __syncthreads();
    for (int s = 128; s > 0; s >>= 1) {
        if (t < s) red[t] += red[t + s];
        __syncthreads();
    }
    if (t == 0) atomicAdd(loss_acc, red[0]);
}

// ---------------------------------------------------------------------------
__global__ void finalize(const float* __restrict__ loss_acc, float* __restrict__ out_loss) {
    *out_loss = 1.25f * (*loss_acc) * (1.0f / (float)ZQ_ELEMS);
}

// ---------------------------------------------------------------------------
extern "C" void kernel_launch(void* const* d_in, const int* in_sizes, int n_in,
                              void* d_out, int out_size, void* d_ws, size_t ws_size,
                              hipStream_t stream) {
    const float* z   = (const float*)d_in[0];   // 16*256*64*64
    const float* emb = (const float*)d_in[1];   // 1024*256
    float* out = (float*)d_out;

    char* ws = (char*)d_ws;
    size_t o = 0;
    unsigned short* zf       = (unsigned short*)(ws + o); o += (size_t)NROWS * C_DIM * 2;  // 32 MB
    unsigned short* wn       = (unsigned short*)(ws + o); o += (size_t)KCODES * C_DIM * 2; // 512 KB
    float*          half_wsq = (float*)(ws + o);          o += (size_t)KCODES * 4;
    float*          inv_norm = (float*)(ws + o);          o += (size_t)B_DIM * C_DIM * H_DIM * 4; // 1 MB
    int*            ws_idx   = (int*)(ws + o);            o += (size_t)NROWS * 4;          // 256 KB
    float*          loss_acc = (float*)(ws + o);          o += 256;

    float* out_zq   = out;
    float* out_idx  = out + ZQ_ELEMS;
    float* out_loss = out + ZQ_ELEMS + NROWS;

    init_acc<<<1, 1, 0, stream>>>(loss_acc);
    prep_w<<<KCODES, 256, 0, stream>>>(emb, wn, half_wsq);
    prep_z<<<B_DIM * H_DIM * (C_DIM / 64), 256, 0, stream>>>(z, zf, inv_norm);
    gemm_argmax<<<NROWS / 64, 128, 0, stream>>>(zf, wn, half_wsq, ws_idx, out_idx);
    epilogue<<<B_DIM * H_DIM * (C_DIM / 64), 256, 0, stream>>>(z, emb, inv_norm, ws_idx,
                                                               out_zq, loss_acc);
    finalize<<<1, 1, 0, stream>>>(loss_acc, out_loss);
}
